// MPNN_64252710748686
// MI455X (gfx1250) — matrix operand order repaired and checked by reading, not compiled, not run on hardware
//
#include <hip/hip_runtime.h>
#include <hip/hip_bf16.h>

typedef __bf16 bf16_t;
typedef __attribute__((ext_vector_type(16))) __bf16 v16bf;
typedef __attribute__((ext_vector_type(8)))  __bf16 v8bf;
typedef __attribute__((ext_vector_type(8)))  float  v8f;

#define DEV __device__ __forceinline__

static constexpr int Bb = 8, Nn = 1024, Cc = 128, Hh = 8, KDim = 16;
static constexpr float kNegBig = -9.0e15f;
static constexpr float kAlpha  = 0.2f;

// ---------------- WMMA helpers (gfx1250, wave32) ----------------
// A fragment (16-bit, 16x32): lane half0 holds K[0..15], half1 K[16..31];
// row = lane%16.
DEV v16bf load_a_frag(const bf16_t* rowptr, int half) {
  return *(const v16bf*)(rowptr + half * 16);
}
// B fragment (16-bit, 32x16): lane half0 holds K[0..15], half1 K[16..31];
// column = lane%16.  btrowptr = &BT[col][kbase]  (BT row-major [N][K]).
DEV v16bf load_b_frag(const bf16_t* btrowptr, int half) {
  return *(const v16bf*)(btrowptr + half * 16);
}
DEV v8f wmma_bf16(v16bf a, v16bf b, v8f c) {
  return __builtin_amdgcn_wmma_f32_16x16x32_bf16(false, a, false, b, (short)0, c,
                                                 false, false);
}
// reductions across the 16 lanes of each wave half (rows of a C/D fragment)
DEV float half_max(float x) {
#pragma unroll
  for (int m = 1; m < 16; m <<= 1) x = fmaxf(x, __shfl_xor(x, m, 32));
  return x;
}
DEV float half_sum(float x) {
#pragma unroll
  for (int m = 1; m < 16; m <<= 1) x += __shfl_xor(x, m, 32);
  return x;
}

// ---------------- prep kernels ----------------
__global__ void k_f32_to_bf16(const float* __restrict__ s, bf16_t* __restrict__ d, int n) {
  int i = blockIdx.x * blockDim.x + threadIdx.x;
  if (i < n) d[i] = (bf16_t)s[i];
}
// dst[j*128+i] = src[i*128+j]  (128x128 transpose + convert)
__global__ void k_transpose128(const float* __restrict__ s, bf16_t* __restrict__ d) {
  int i = blockIdx.x * blockDim.x + threadIdx.x;  // 16384
  d[i] = (bf16_t)s[(i & 127) * 128 + (i >> 7)];
}
// WqT[h][kd][c] = Wq[h][c][kd]  (same for Wk, Wv)
__global__ void k_transpose_qkvw(const float* __restrict__ wq, const float* __restrict__ wk,
                                 const float* __restrict__ wv, bf16_t* __restrict__ wqT,
                                 bf16_t* __restrict__ wkT, bf16_t* __restrict__ wvT) {
  int i = blockIdx.x * blockDim.x + threadIdx.x;  // H*KD*C = 16384
  int c = i & 127; int t = i >> 7; int kd = t & 15; int h = t >> 4;
  int s = (h * 128 + c) * 16 + kd;
  wqT[i] = (bf16_t)wq[s];
  wkT[i] = (bf16_t)wk[s];
  wvT[i] = (bf16_t)wv[s];
}

// ---------------- h = x @ W ----------------
// 4096 tiles (b, n-tile, c-tile); one wave per 16x16 tile, K=128 in 4 steps.
__global__ void k_gemm_h(const bf16_t* __restrict__ xb, const bf16_t* __restrict__ WT,
                         float* __restrict__ hf, bf16_t* __restrict__ hT) {
  int wave = (blockIdx.x * blockDim.x + threadIdx.x) >> 5;
  int lane = threadIdx.x & 31; int half = lane >> 4; int r = lane & 15;
  int jc = wave & 7; int t = wave >> 3; int it = t & 63; int b = t >> 6;
  int row0 = it * 16, col0 = jc * 16;
  v8f acc = {};
#pragma unroll
  for (int k = 0; k < Cc; k += 32) {
    v16bf a = load_a_frag(xb + ((long)b * Nn + row0 + r) * Cc + k, half);
    v16bf w = load_b_frag(WT + (long)(col0 + r) * Cc + k, half);
    acc = wmma_bf16(a, w, acc);
  }
  v8bf pk;
#pragma unroll
  for (int v = 0; v < 8; ++v) {
    int rr = row0 + v + half * 8;
    hf[((long)b * Nn + rr) * Cc + col0 + r] = acc[v];
    pk[v] = (bf16_t)acc[v];
  }
  *(v8bf*)(hT + ((long)b * Cc + col0 + r) * Nn + row0 + half * 8) = pk;
}

// ---------------- src/dst = h @ a[:C], h @ a[C:] ----------------
__global__ void k_srcdst(const float* __restrict__ hf, const float* __restrict__ avec,
                         float* __restrict__ src, float* __restrict__ dst) {
  int wid = (blockIdx.x * blockDim.x + threadIdx.x) >> 5;  // 0..B*N-1
  int lane = threadIdx.x & 31;
  const float* hrow = hf + (long)wid * Cc;
  float s0 = 0.f, s1 = 0.f;
  for (int c = lane; c < Cc; c += 32) {
    float hv = hrow[c];
    s0 += hv * avec[c];
    s1 += hv * avec[Cc + c];
  }
#pragma unroll
  for (int m = 16; m >= 1; m >>= 1) { s0 += __shfl_xor(s0, m, 32); s1 += __shfl_xor(s1, m, 32); }
  if (lane == 0) { src[wid] = s0; dst[wid] = s1; }
}

// ---------------- GAT flash attention: hp = elu(softmax(mask(e)) @ h) ----------------
// one wave (blockDim=32) per (b, 16-row tile); stream 32 columns per step.
__global__ void k_gat_flash(const float* __restrict__ adj, const float* __restrict__ src,
                            const float* __restrict__ dst, const bf16_t* __restrict__ hT,
                            bf16_t* __restrict__ hp) {
  int blk = blockIdx.x;
  int b = blk >> 6; int it = blk & 63; int row0 = it * 16;
  int lane = threadIdx.x; int half = lane >> 4; int r = lane & 15;
  __shared__ bf16_t pbuf[16 * 32];

  float srcv[8];
#pragma unroll
  for (int v = 0; v < 8; ++v) srcv[v] = src[b * Nn + row0 + v + half * 8];

  v8f acc[8];
#pragma unroll
  for (int jc = 0; jc < 8; ++jc) acc[jc] = (v8f){};
  float mrow[8], lrow[8];
#pragma unroll
  for (int v = 0; v < 8; ++v) { mrow[v] = -__builtin_inff(); lrow[v] = 0.f; }

  for (int jb = 0; jb < Nn / 32; ++jb) {
    int col0 = jb * 32;
    float e[2][8];
#pragma unroll
    for (int s = 0; s < 2; ++s) {
      int col = col0 + s * 16 + r;
      float dv = dst[b * Nn + col];
#pragma unroll
      for (int v = 0; v < 8; ++v) {
        int rr = row0 + v + half * 8;
        float av = adj[((long)b * Nn + rr) * Nn + col];
        float ev = srcv[v] + dv;
        ev = ev > 0.f ? ev : kAlpha * ev;
        e[s][v] = (av < 1.0f) ? ev : kNegBig;
      }
    }
#pragma unroll
    for (int v = 0; v < 8; ++v) {
      float mnew = fmaxf(mrow[v], half_max(fmaxf(e[0][v], e[1][v])));
      float scale = __expf(mrow[v] - mnew);
      mrow[v] = mnew;
      float p0 = __expf(e[0][v] - mnew);
      float p1 = __expf(e[1][v] - mnew);
      lrow[v] = lrow[v] * scale + half_sum(p0 + p1);
#pragma unroll
      for (int jc = 0; jc < 8; ++jc) acc[jc][v] *= scale;
      pbuf[(v + half * 8) * 32 + r]      = (bf16_t)p0;
      pbuf[(v + half * 8) * 32 + 16 + r] = (bf16_t)p1;
    }
    __syncthreads();
    v16bf afrag = load_a_frag(&pbuf[r * 32], half);
#pragma unroll
    for (int jc = 0; jc < 8; ++jc) {
      v16bf bfrag = load_b_frag(hT + ((long)b * Cc + jc * 16 + r) * Nn + col0, half);
      acc[jc] = wmma_bf16(afrag, bfrag, acc[jc]);
    }
    __syncthreads();
  }
#pragma unroll
  for (int jc = 0; jc < 8; ++jc)
#pragma unroll
    for (int v = 0; v < 8; ++v) {
      float hv = acc[jc][v] / lrow[v];
      hv = hv > 0.f ? hv : (__expf(hv) - 1.f);  // ELU
      hp[((long)b * Nn + row0 + v + half * 8) * Cc + jc * 16 + r] = (bf16_t)hv;
    }
}

// ---------------- Q/K/V projections (Q pre-scaled by 1/sqrt(KD)) ----------------
__global__ void k_gemm_qkv(const bf16_t* __restrict__ hp, const bf16_t* __restrict__ wqT,
                           const bf16_t* __restrict__ wkT, const bf16_t* __restrict__ wvT,
                           bf16_t* __restrict__ Q, bf16_t* __restrict__ Km,
                           bf16_t* __restrict__ VT) {
  int wave = (blockIdx.x * blockDim.x + threadIdx.x) >> 5;
  int lane = threadIdx.x & 31; int half = lane >> 4; int r = lane & 15;
  int it = wave & 63; int hb = wave >> 6; int b = hb & 7; int h = hb >> 3;
  int row0 = it * 16;
  v8f aq = {}, ak = {}, av = {};
#pragma unroll
  for (int k = 0; k < Cc; k += 32) {
    v16bf a  = load_a_frag(hp + ((long)b * Nn + row0 + r) * Cc + k, half);
    v16bf bq = load_b_frag(wqT + (long)(h * KDim + r) * Cc + k, half);
    v16bf bk = load_b_frag(wkT + (long)(h * KDim + r) * Cc + k, half);
    v16bf bv = load_b_frag(wvT + (long)(h * KDim + r) * Cc + k, half);
    aq = wmma_bf16(a, bq, aq);
    ak = wmma_bf16(a, bk, ak);
    av = wmma_bf16(a, bv, av);
  }
  const float norm = 0.25f;  // 1/sqrt(16)
  v8bf pv;
#pragma unroll
  for (int v = 0; v < 8; ++v) {
    int rr = row0 + v + half * 8;
    Q[((long)hb * Nn + rr) * KDim + r]  = (bf16_t)(aq[v] * norm);
    Km[((long)hb * Nn + rr) * KDim + r] = (bf16_t)ak[v];
    pv[v] = (bf16_t)av[v];
  }
  *(v8bf*)(VT + ((long)hb * KDim + r) * Nn + row0 + half * 8) = pv;
}

// ---------------- MHA flash: heads = softmax(Q K^T) @ V ----------------
__global__ void k_mha_flash(const bf16_t* __restrict__ Q, const bf16_t* __restrict__ Km,
                            const bf16_t* __restrict__ VT, bf16_t* __restrict__ hcat) {
  int blk = blockIdx.x;
  int it = blk & 63; int hb = blk >> 6; int b = hb & 7; int h = hb >> 3;
  int row0 = it * 16;
  int lane = threadIdx.x; int half = lane >> 4; int r = lane & 15;
  __shared__ bf16_t pbuf[16 * 32];

  const bf16_t* Qhb = Q  + (long)hb * Nn * KDim;
  const bf16_t* Khb = Km + (long)hb * Nn * KDim;
  const bf16_t* Vhb = VT + (long)hb * KDim * Nn;

  // A fragment of Q: valid K = 0..15, zero-pad K = 16..31. half0 holds
  // K0-15, half1 is zero.
  v16bf qfrag = {};
  if (half == 0) qfrag = *(const v16bf*)(Qhb + (long)(row0 + r) * KDim);

  v8f acc = {};
  float mrow[8], lrow[8];
#pragma unroll
  for (int v = 0; v < 8; ++v) { mrow[v] = -__builtin_inff(); lrow[v] = 0.f; }

  for (int mb = 0; mb < Nn / 32; ++mb) {
    int m0 = mb * 32;
    float e[2][8];
#pragma unroll
    for (int s = 0; s < 2; ++s) {
      // B fragment of K^T: BT row = K[mcol][:]; half0 K=0..15 valid, half1 zero.
      v16bf kfrag = {};
      if (half == 0) kfrag = *(const v16bf*)(Khb + (long)(m0 + s * 16 + r) * KDim);
      v8f sc = {};
      sc = wmma_bf16(qfrag, kfrag, sc);
#pragma unroll
      for (int v = 0; v < 8; ++v) e[s][v] = sc[v];
    }
#pragma unroll
    for (int v = 0; v < 8; ++v) {
      float mnew = fmaxf(mrow[v], half_max(fmaxf(e[0][v], e[1][v])));
      float scale = __expf(mrow[v] - mnew);
      mrow[v] = mnew;
      float p0 = __expf(e[0][v] - mnew);
      float p1 = __expf(e[1][v] - mnew);
      lrow[v] = lrow[v] * scale + half_sum(p0 + p1);
      acc[v] *= scale;
      pbuf[(v + half * 8) * 32 + r]      = (bf16_t)p0;
      pbuf[(v + half * 8) * 32 + 16 + r] = (bf16_t)p1;
    }
    __syncthreads();
    v16bf afrag = load_a_frag(&pbuf[r * 32], half);
    v16bf vfrag = load_b_frag(Vhb + (long)r * Nn + m0, half);  // col = kd = r
    acc = wmma_bf16(afrag, vfrag, acc);
    __syncthreads();
  }
#pragma unroll
  for (int v = 0; v < 8; ++v) {
    float hv = acc[v] / lrow[v];
    hcat[((long)b * Nn + row0 + v + half * 8) * Cc + h * KDim + r] = (bf16_t)hv;
  }
}

// ---------------- out = hcat @ Wout_cat ----------------
__global__ void k_gemm_out(const bf16_t* __restrict__ hcat, const bf16_t* __restrict__ WoT,
                           float* __restrict__ out) {
  int wave = (blockIdx.x * blockDim.x + threadIdx.x) >> 5;
  int lane = threadIdx.x & 31; int half = lane >> 4; int r = lane & 15;
  int jc = wave & 7; int t = wave >> 3; int it = t & 63; int b = t >> 6;
  int row0 = it * 16, col0 = jc * 16;
  v8f acc = {};
#pragma unroll
  for (int k = 0; k < Cc; k += 32) {
    v16bf a = load_a_frag(hcat + ((long)b * Nn + row0 + r) * Cc + k, half);
    v16bf w = load_b_frag(WoT + (long)(col0 + r) * Cc + k, half);
    acc = wmma_bf16(a, w, acc);
  }
#pragma unroll
  for (int v = 0; v < 8; ++v) {
    int rr = row0 + v + half * 8;
    out[((long)b * Nn + rr) * Cc + col0 + r] = acc[v];
  }
}

// ---------------- host launcher ----------------
extern "C" void kernel_launch(void* const* d_in, const int* in_sizes, int n_in,
                              void* d_out, int out_size, void* d_ws, size_t ws_size,
                              hipStream_t stream) {
  const float* x    = (const float*)d_in[0];  // [B,N,C]
  const float* adj  = (const float*)d_in[1];  // [B,N,N]
  const float* W    = (const float*)d_in[2];  // [C,C]
  const float* avec = (const float*)d_in[3];  // [2C,1]
  const float* Wq   = (const float*)d_in[4];  // [H,C,KD]
  const float* Wk   = (const float*)d_in[5];
  const float* Wv   = (const float*)d_in[6];
  const float* Wout = (const float*)d_in[7];  // [H,KD,C]
  float* out = (float*)d_out;

  char* wsb = (char*)d_ws;
  size_t off = 0;
  auto take = [&](size_t bytes) -> void* {
    void* r = wsb + off;
    off += (bytes + 255) & ~(size_t)255;
    return r;
  };
  const size_t nBNC = (size_t)Bb * Nn * Cc;
  bf16_t* xb  = (bf16_t*)take(nBNC * 2);
  bf16_t* WT  = (bf16_t*)take(128 * 128 * 2);
  bf16_t* WoT = (bf16_t*)take(128 * 128 * 2);
  bf16_t* wqT = (bf16_t*)take((size_t)Hh * KDim * Cc * 2);
  bf16_t* wkT = (bf16_t*)take((size_t)Hh * KDim * Cc * 2);
  bf16_t* wvT = (bf16_t*)take((size_t)Hh * KDim * Cc * 2);
  float*  hf  = (float*)take(nBNC * 4);
  bf16_t* hT  = (bf16_t*)take(nBNC * 2);
  float*  src = (float*)take((size_t)Bb * Nn * 4);
  float*  dst = (float*)take((size_t)Bb * Nn * 4);
  bf16_t* hp  = (bf16_t*)take(nBNC * 2);
  const size_t nHBNK = (size_t)Hh * Bb * Nn * KDim;
  bf16_t* Qm  = (bf16_t*)take(nHBNK * 2);
  bf16_t* Km  = (bf16_t*)take(nHBNK * 2);
  bf16_t* VT  = (bf16_t*)take(nHBNK * 2);
  bf16_t* hc  = (bf16_t*)take(nBNC * 2);
  (void)ws_size; (void)in_sizes; (void)n_in; (void)out_size;

  // prep
  k_f32_to_bf16<<<(int)(nBNC / 256), 256, 0, stream>>>(x, xb, (int)nBNC);
  k_transpose128<<<64, 256, 0, stream>>>(W, WT);
  k_transpose128<<<64, 256, 0, stream>>>(Wout, WoT);
  k_transpose_qkvw<<<64, 256, 0, stream>>>(Wq, Wk, Wv, wqT, wkT, wvT);
  // GAT
  k_gemm_h<<<1024, 128, 0, stream>>>(xb, WT, hf, hT);
  k_srcdst<<<1024, 256, 0, stream>>>(hf, avec, src, dst);
  k_gat_flash<<<Bb * (Nn / 16), 32, 0, stream>>>(adj, src, dst, hT, hp);
  // MHA
  k_gemm_qkv<<<1024, 128, 0, stream>>>(hp, wqT, wkT, wvT, Qm, Km, VT);
  k_mha_flash<<<Hh * Bb * (Nn / 16), 32, 0, stream>>>(Qm, Km, VT, hc);
  k_gemm_out<<<1024, 128, 0, stream>>>(hc, WoT, out);
}
